// Model_12266426597389
// MI455X (gfx1250) — matrix operand order repaired and checked by reading, not compiled, not run on hardware
//
#include <hip/hip_runtime.h>

// ---------------------------------------------------------------------------
// GNN pipeline for MI455X (gfx1250, wave32, WMMA).
//   GEMMs via v_wmma_f32_16x16x32_f16 (f32 accumulate), A staged to LDS with
//   gfx1250 global_load_async_to_lds_b128 (double-buffered, ASYNCcnt waits),
//   B (weights) pre-transposed once and held in LDS for the whole block.
//   SpMM scatter via float4 gathers + global_atomic_add_f32 (L2-resident).
// ---------------------------------------------------------------------------

typedef __attribute__((ext_vector_type(16))) _Float16 v16h;
typedef __attribute__((ext_vector_type(8)))  float    v8f;

#define THREADS 256
#define USE_ASYNC_COPY 1   // gfx1250 async global->LDS path (flip to 0 for sync fallback)

// ------------------------------- utility kernels ---------------------------

__global__ void k_zero(float* __restrict__ p, long n) {
    long i = (long)blockIdx.x * THREADS + threadIdx.x;
    if (i < n) p[i] = 0.0f;
}

__global__ void k_cvt_f16(const float* __restrict__ in, _Float16* __restrict__ out, long n) {
    long i = (long)blockIdx.x * THREADS + threadIdx.x;
    if (i < n) out[i] = (_Float16)in[i];
}

// Wt[n, k] = f16(W[k, n]);  W: [K, NCOLS] f32 row-major -> Wt: [NCOLS, K] f16
__global__ void k_cvt_transpose(const float* __restrict__ W, _Float16* __restrict__ Wt,
                                int K, int Ncols) {
    long idx = (long)blockIdx.x * THREADS + threadIdx.x;
    if (idx >= (long)K * Ncols) return;
    const int n = (int)(idx / K);
    const int k = (int)(idx % K);
    Wt[idx] = (_Float16)W[(size_t)k * Ncols + n];
}

__global__ void k_degree(const int* __restrict__ src, const int* __restrict__ dst,
                         float* __restrict__ degO, float* __restrict__ degI, int E) {
    int e = blockIdx.x * THREADS + threadIdx.x;
    if (e < E) {
        atomicAdd(&degO[src[e]], 1.0f);
        atomicAdd(&degI[dst[e]], 1.0f);
    }
}

__global__ void k_norm(const float* __restrict__ degO, const float* __restrict__ degI,
                       float* __restrict__ nO, float* __restrict__ nI, int n) {
    int i = blockIdx.x * THREADS + threadIdx.x;
    if (i < n) {
        float a = degO[i];
        float b = degI[i];
        nO[i] = (a > 0.0f) ? rsqrtf(a) : 0.0f;
        nI[i] = (b > 0.0f) ? rsqrtf(b) : 0.0f;
    }
}

// ------------------------------- WMMA GEMM ---------------------------------
// C[M, NCOLS] = A[M, 256](f16) @ Bt[NCOLS, 256](f16, pre-transposed)
//               (+bias) (*rowscale)
// block: 256 threads = 8 waves; macro tile 128 rows x 64 cols.
// B panel (64 x 256) lives in LDS for the whole block; A double-buffered via
// async global->LDS copies.

template <int NCOLS, bool HALF_OUT>
__global__ __launch_bounds__(THREADS)
void k_gemm_wmma(const _Float16* __restrict__ A, const _Float16* __restrict__ Bt,
                 const float* __restrict__ bias, const float* __restrict__ rowscale,
                 float* __restrict__ Cf, _Float16* __restrict__ Ch, int M) {
    constexpr int K   = 256;
    constexpr int AST = 40;       // A tile row stride (halves): 80 B, 16 B aligned
    constexpr int BST = K + 8;    // B panel row stride (halves): 528 B -> bank-spread

    __shared__ _Float16 As[2][128 * AST];   // ping-pong 128 rows x 32 k  (2 x 10 KB)
    __shared__ _Float16 Bs[64 * BST];       // [col][k] full-K panel      (33 KB)

    const int tid      = threadIdx.x;
    const int wave     = tid >> 5;
    const int lane     = tid & 31;
    const int rowBlock = blockIdx.x * 128;
    const int colBase  = blockIdx.y * 64;
    const int waveRow  = rowBlock + wave * 16;

    // ---- load entire B panel once (pure vector copies, no transpose here) ----
    {
        const int r   = tid >> 2;           // 0..63 : output column within panel
        const int seg = (tid & 3) * 64;     // halves along K
        const float4* gp = (const float4*)(Bt + (size_t)(colBase + r) * K + seg);
        float4*       sp = (float4*)(&Bs[r * BST + seg]);
#pragma unroll
        for (int i = 0; i < 8; ++i) sp[i] = gp[i];
    }

    // ---- A staging: each thread moves 16 halves (two 16 B beats) of one row ----
    const int arow  = tid >> 1;
    const int akoff = (tid & 1) * 16;
    const int grow  = rowBlock + arow;

    auto stageA = [&](int buf, int k0) {
        if (grow < M) {
            const _Float16* gp = A + (size_t)grow * K + k0 + akoff;
#if USE_ASYNC_COPY
            // INST_OFFSET is added to both the global and the LDS address, so
            // one base covers both 16 B beats. LDS offset = low 32 bits of the
            // generic pointer (wave-relative).
            const unsigned lp = (unsigned)(uintptr_t)(&As[buf][arow * AST + akoff]);
            asm volatile("global_load_async_to_lds_b128 %0, %1, off"
                         :: "v"(lp), "v"(gp) : "memory");
            asm volatile("global_load_async_to_lds_b128 %0, %1, off offset:16"
                         :: "v"(lp), "v"(gp) : "memory");
#else
            const float4* g4 = (const float4*)gp;
            float4*       sp = (float4*)(&As[buf][arow * AST + akoff]);
            sp[0] = g4[0];
            sp[1] = g4[1];
            __builtin_prefetch((const void*)(gp + 64), 0, 1);
#endif
        }
#if !USE_ASYNC_COPY
        else {
            float4* sp = (float4*)(&As[buf][arow * AST + akoff]);
            sp[0] = make_float4(0.f, 0.f, 0.f, 0.f);
            sp[1] = make_float4(0.f, 0.f, 0.f, 0.f);
        }
#endif
    };

    // fragment addressing (documented wave32 VGPR layouts)
    const int mrow = lane & 15;
    const int kb   = (lane >> 4) * 8;    // A: K 0-7/16-23 vs 8-15/24-31
    const int kb2  = (lane >> 4) * 16;   // unused; B fragments also use kb (same k order as A)

    union FU { float4 f[2]; v16h v; };
    v8f acc[4] = {};

    // prologue: fill buffer 0
    stageA(0, 0);
#if USE_ASYNC_COPY
    asm volatile("s_wait_asynccnt 0x0" ::: "memory");
#endif
    __syncthreads();

    int buf = 0;
    for (int k0 = 0; k0 < K; k0 += 32, buf ^= 1) {
        if (k0 + 32 < K) stageA(buf ^ 1, k0 + 32);   // async fill of next tile

        FU af;
        af.f[0] = *(const float4*)(&As[buf][(wave * 16 + mrow) * AST + kb]);
        af.f[1] = *(const float4*)(&As[buf][(wave * 16 + mrow) * AST + 16 + kb]);

        // Manual unroll so the reuse_a modifier is a literal constant:
        // A fragment is shared by the following WMMA -> reuse_a=1 on tiles 0..2.
#define WMMA_TILE(T, RA)                                                        \
        {                                                                       \
            FU bf;                                                              \
            const int bcol = (T) * 16 + mrow;                                   \
            bf.f[0] = *(const float4*)(&Bs[bcol * BST + k0 + kb]);              \
            bf.f[1] = *(const float4*)(&Bs[bcol * BST + k0 + kb + 16]);         \
            acc[T] = __builtin_amdgcn_wmma_f32_16x16x32_f16(                    \
                false, af.v, false, bf.v, (short)0, acc[T], (RA), false);       \
        }
        WMMA_TILE(0, true)
        WMMA_TILE(1, true)
        WMMA_TILE(2, true)
        WMMA_TILE(3, false)
#undef WMMA_TILE

#if USE_ASYNC_COPY
        asm volatile("s_wait_asynccnt 0x0" ::: "memory");
#endif
        __syncthreads();
    }

    if (waveRow >= M) return;   // uniform per wave: EXEC stays all-ones elsewhere

    // C/D layout: VGPR r -> row r (lanes 0-15) / row 8+r (lanes 16-31), col = lane&15
    const int colLane = colBase + (lane & 15);
    const int rAdd    = (lane >> 4) * 8;
#pragma unroll
    for (int t = 0; t < 4; ++t) {
        const int col = colLane + t * 16;
        const float bv = bias ? bias[col] : 0.0f;
#pragma unroll
        for (int r = 0; r < 8; ++r) {
            const int row = waveRow + rAdd + r;
            const float v = acc[t][r] + bv;
            if constexpr (HALF_OUT) {
                Ch[(size_t)row * NCOLS + col] = (_Float16)(v * rowscale[row]);
            } else {
                Cf[(size_t)row * NCOLS + col] = v;
            }
        }
    }
}

// ------------------------------- SpMM scatter ------------------------------
// agg[dst[e], :] += y[src[e], :]   (NC/4 threads per edge, float4 gathers)

template <int NC>
__global__ void k_scatter(const float* __restrict__ y, const int* __restrict__ src,
                          const int* __restrict__ dst, float* __restrict__ agg, int E) {
    constexpr int CH = NC / 4;
    long idx = (long)blockIdx.x * THREADS + threadIdx.x;
    if (idx >= (long)E * CH) return;
    const int e = (int)(idx / CH);
    const int c = (int)(idx % CH) * 4;
    const int s = src[e];
    const int d = dst[e];
    const float4 v = *(const float4*)(y + (size_t)s * NC + c);
    float* ap = agg + (size_t)d * NC + c;
    atomicAdd(ap + 0, v.x);
    atomicAdd(ap + 1, v.y);
    atomicAdd(ap + 2, v.z);
    atomicAdd(ap + 3, v.w);
}

// h1 = relu(agg*norm_in + b1); emit f16 of h1*norm_out (next GEMM's A operand)
__global__ void k_h1(const float* __restrict__ agg, const float* __restrict__ nI,
                     const float* __restrict__ nO, const float* __restrict__ b,
                     _Float16* __restrict__ xh, int N) {
    long idx = (long)blockIdx.x * THREADS + threadIdx.x;
    if (idx >= (long)N * 256) return;
    const int n = (int)(idx >> 8);
    const int c = (int)(idx & 255);
    float v = fmaxf(agg[idx] * nI[n] + b[c], 0.0f);
    xh[idx] = (_Float16)(v * nO[n]);
}

// h2 = relu(agg*norm_in + b2) in f32
__global__ void k_h2(const float* __restrict__ agg, const float* __restrict__ nI,
                     const float* __restrict__ b, float* __restrict__ h2, int N) {
    long idx = (long)blockIdx.x * THREADS + threadIdx.x;
    if (idx >= (long)N * 128) return;
    const int n = (int)(idx >> 7);
    const int c = (int)(idx & 127);
    h2[idx] = fmaxf(agg[idx] * nI[n] + b[c], 0.0f);
}

// score[e] = dot(h2[src[e]], h2[dst[e]]) over 128 dims; 4 lanes per edge
__global__ void k_score(const float* __restrict__ h2, const int* __restrict__ src,
                        const int* __restrict__ dst, float* __restrict__ out, int E) {
    long idx = (long)blockIdx.x * THREADS + threadIdx.x;
    int e = (int)(idx >> 2);
    const int part = (int)(idx & 3);
    const bool valid = (e < E);
    if (e >= E) e = E - 1;           // keep wave-uniform shfl participation
    const int s = src[e];
    const int d = dst[e];
    const float4* a = (const float4*)(h2 + (size_t)s * 128 + part * 32);
    const float4* b = (const float4*)(h2 + (size_t)d * 128 + part * 32);
    float acc = 0.0f;
#pragma unroll
    for (int i = 0; i < 8; ++i) {
        const float4 x = a[i];
        const float4 y = b[i];
        acc += x.x * y.x + x.y * y.y + x.z * y.z + x.w * y.w;
    }
    acc += __shfl_xor(acc, 1);
    acc += __shfl_xor(acc, 2);
    if (valid && part == 0) out[e] = acc;
}

// ------------------------------- launcher ----------------------------------

static inline dim3 g1(long n) { return dim3((unsigned)((n + THREADS - 1) / THREADS)); }

extern "C" void kernel_launch(void* const* d_in, const int* in_sizes, int n_in,
                              void* d_out, int out_size, void* d_ws, size_t ws_size,
                              hipStream_t stream) {
    const float* X    = (const float*)d_in[0];
    // d_in[1] = edge_features: unused by the reference math
    const int*   src  = (const int*)d_in[2];
    const int*   dst  = (const int*)d_in[3];
    const float* Wlin = (const float*)d_in[4];
    const float* blin = (const float*)d_in[5];
    const float* W1   = (const float*)d_in[6];
    const float* b1   = (const float*)d_in[7];
    const float* W2   = (const float*)d_in[8];
    const float* b2   = (const float*)d_in[9];
    float* out = (float*)d_out;

    const int N = in_sizes[0] / 256;   // 50000 (divisible by 16)
    const int E = in_sizes[2];         // 1600000

    // workspace carve-out (aliased: ~155 MB total)
    char* w = (char*)d_ws;
    auto carve = [&](size_t bytes) { char* p = w; w += (bytes + 255) & ~(size_t)255; return p; };
    float*    degO  = (float*)carve((size_t)N * 4);
    float*    degI  = (float*)carve((size_t)N * 4);
    float*    nO    = (float*)carve((size_t)N * 4);
    float*    nI    = (float*)carve((size_t)N * 4);
    _Float16* WlinT = (_Float16*)carve(256 * 256 * 2);  // [256 out, 256 k]
    _Float16* W1T   = (_Float16*)carve(256 * 256 * 2);  // [256 out, 256 k]
    _Float16* W2T   = (_Float16*)carve(128 * 256 * 2);  // [128 out, 256 k]
    _Float16* HA    = (_Float16*)carve((size_t)N * 256 * 2); // Xh, later xh2
    _Float16* HB    = (_Float16*)carve((size_t)N * 256 * 2); // xh1
    float*    FA    = (float*)carve((size_t)N * 256 * 4);    // y1; later y2 | h2
    float*    FB    = (float*)carve((size_t)N * 256 * 4);    // agg1; later agg2

    const dim3 blk(THREADS);

    // degrees + norms
    k_zero<<<g1(N), blk, 0, stream>>>(degO, N);
    k_zero<<<g1(N), blk, 0, stream>>>(degI, N);
    k_degree<<<g1(E), blk, 0, stream>>>(src, dst, degO, degI, E);
    k_norm<<<g1(N), blk, 0, stream>>>(degO, degI, nO, nI, N);

    // f16 conversions: activations plain, weights transposed to [out, k]
    k_cvt_f16<<<g1((long)N * 256), blk, 0, stream>>>(X, HA, (long)N * 256);
    k_cvt_transpose<<<g1(256 * 256), blk, 0, stream>>>(Wlin, WlinT, 256, 256);
    k_cvt_transpose<<<g1(256 * 256), blk, 0, stream>>>(W1, W1T, 256, 256);
    k_cvt_transpose<<<g1(256 * 128), blk, 0, stream>>>(W2, W2T, 256, 128);

    const unsigned mblocks = (unsigned)((N + 127) / 128);

    // GEMM1: xh1 = f16( (X@Wlin + blin) * norm_out )
    k_gemm_wmma<256, true><<<dim3(mblocks, 4), blk, 0, stream>>>(
        HA, WlinT, blin, nO, nullptr, HB, N);

    // GEMM2: y1 = xh1 @ W1
    k_gemm_wmma<256, false><<<dim3(mblocks, 4), blk, 0, stream>>>(
        HB, W1T, nullptr, nullptr, FA, nullptr, N);

    // conv1 aggregate + activation
    k_zero<<<g1((long)N * 256), blk, 0, stream>>>(FB, (long)N * 256);
    k_scatter<256><<<g1((long)E * 64), blk, 0, stream>>>(FA, src, dst, FB, E);
    k_h1<<<g1((long)N * 256), blk, 0, stream>>>(FB, nI, nO, b1, HA, N);  // HA := xh2

    // GEMM3: y2 = xh2 @ W2
    k_gemm_wmma<128, false><<<dim3(mblocks, 2), blk, 0, stream>>>(
        HA, W2T, nullptr, nullptr, FA, nullptr, N);

    // conv2 aggregate + activation
    k_zero<<<g1((long)N * 128), blk, 0, stream>>>(FB, (long)N * 128);
    k_scatter<128><<<g1((long)E * 32), blk, 0, stream>>>(FA, src, dst, FB, E);
    float* h2 = FA + (size_t)N * 128;
    k_h2<<<g1((long)N * 128), blk, 0, stream>>>(FB, nI, b2, h2, N);

    // per-edge dot-product scores
    k_score<<<g1((long)E * 4), blk, 0, stream>>>(h2, src, dst, out, E);
}
